// MultiHeadAttention_7584912244958
// MI455X (gfx1250) — matrix operand order repaired and checked by reading, not compiled, not run on hardware
//
#include <hip/hip_runtime.h>

typedef __attribute__((ext_vector_type(16))) _Float16 v16h;
typedef __attribute__((ext_vector_type(8)))  _Float16 v8h;
typedef __attribute__((ext_vector_type(8)))  float    v8f;

#define D_MODEL 512
#define SEQ     4096
#define BATCH   2
#define HEADS   8
#define DK      64
#define M_TOT   (BATCH * SEQ)   // 8192 rows

__device__ __forceinline__ v8f wmma_f16(v16h a, v16h b, v8f c) {
  return __builtin_amdgcn_wmma_f32_16x16x32_f16(
      /*neg_a=*/false, a, /*neg_b=*/false, b,
      /*c_mod=*/(short)0, c, /*reuse_a=*/false, /*reuse_b=*/false);
}

// ---------------------------------------------------------------------------
// Tensor Data Mover (guarded): 2D tile load Global->LDS with HW LDS padding.
// Descriptor bit layout per CDNA5 ISA ch.8 (D# groups 0/1).
// ---------------------------------------------------------------------------
#if defined(__has_builtin)
#if __has_builtin(__builtin_amdgcn_tensor_load_to_lds) && \
    __has_builtin(__builtin_amdgcn_s_wait_tensorcnt)
#define HAVE_TDM 1
#endif
#endif

#ifdef HAVE_TDM
#pragma message("CDNA5: TDM tensor_load_to_lds path ENABLED")
#else
#pragma message("CDNA5: TDM builtins NOT FOUND - cooperative-copy fallback in use")
#endif

#ifdef HAVE_TDM
typedef unsigned int uint32x4 __attribute__((ext_vector_type(4)));
typedef int          int32x8  __attribute__((ext_vector_type(8)));
typedef int          int32x4  __attribute__((ext_vector_type(4)));

__device__ __forceinline__ void tdm_load_2d(
    unsigned int lds_addr, const void* gptr,
    unsigned int tensor_d0_dw,   // tensor_dim0 (dwords)
    unsigned int tile_d0_dw,     // tile_dim0 (dwords)
    unsigned int tile_rows,      // tile_dim1
    unsigned int tensor_rows,    // tensor_dim1
    unsigned int stride_dw,      // tensor_dim0_stride (dwords)
    unsigned int pad_int_enc, unsigned int pad_amt_enc, unsigned int pad_en)
{
  unsigned long long ga = (unsigned long long)gptr;
  uint32x4 g0;
  g0[0] = 1u;                                       // count=1, user mode
  g0[1] = lds_addr;                                 // lds_addr [63:32]
  g0[2] = (unsigned int)(ga & 0xffffffffull);       // global_addr low
  g0[3] = (unsigned int)((ga >> 32) & 0x1ffffffull) // global_addr high (57b)
          | (2u << 30);                             // type = 2 ("image")
  int32x8 g1;
  g1[0] = (int)((2u << 16)                          // data_size = 4B
                | (pad_en << 20)
                | (pad_int_enc << 22)
                | (pad_amt_enc << 25));
  g1[1] = (int)(tensor_d0_dw << 16);                // tensor_dim0[15:0] @63:48
  g1[2] = (int)((tensor_d0_dw >> 16) | (tensor_rows << 16));
  g1[3] = (int)((tensor_rows >> 16) | (tile_d0_dw << 16));  // tile_dim0 @127:112
  g1[4] = (int)(tile_rows);                         // tile_dim1 (tile_dim2=0)
  g1[5] = (int)(stride_dw);                         // tensor_dim0_stride low32
  g1[6] = 0;
  g1[7] = 0;
  int32x4 z4 = {0, 0, 0, 0};
#if __clang_major__ >= 23
  int32x8 z8 = {0, 0, 0, 0, 0, 0, 0, 0};
  __builtin_amdgcn_tensor_load_to_lds(g0, g1, z4, z4, z8, 0);
#else
  __builtin_amdgcn_tensor_load_to_lds(g0, g1, z4, z4, 0);
#endif
}
#endif

// ---------------------------------------------------------------------------
// Kernel 1: projection  (X @ W^T + bias) -> f16.
// vmode 0: Out[B][H][S][DK]   (Q, K layout)
// vmode 1: Out[B][H][DK][S]   (V transposed: TDM-friendly + packed stores)
// Block = 256 thr computes 128(M) x 64(N); K chunks of 32 staged via LDS.
// ---------------------------------------------------------------------------
__global__ __launch_bounds__(256) void qkv_proj_kernel(
    const float* __restrict__ X, const float* __restrict__ W,
    const float* __restrict__ bias, _Float16* __restrict__ Out, int vmode)
{
  __shared__ _Float16 As[128 * 40];   // stride 40 halfs (80B, 16B-aligned)
  __shared__ _Float16 Bs[64 * 40];

  const int tid  = threadIdx.x;
  const int lane = tid & 31;
  const int wave = tid >> 5;
  const int m0 = (blockIdx.x >> 3) * 128;    // 64 m-blocks
  const int n0 = (blockIdx.x & 7) * 64;      // 8 n-blocks
  const int row = lane & 15;
  const int kbA = (lane >> 4) * 8;
  const int kbB = (lane >> 4) * 16;
  const int col = lane & 15;

  v8f acc[4];
#pragma unroll
  for (int t = 0; t < 4; ++t) acc[t] = (v8f){};

  const float2* __restrict__ X2 = (const float2*)X;
  const float2* __restrict__ W2 = (const float2*)W;

  for (int k0 = 0; k0 < D_MODEL; k0 += 32) {
    __syncthreads();
#pragma unroll
    for (int i = 0; i < 8; ++i) {            // A: 128x32 f32 -> f16
      const int e = tid + i * 256;
      const int r = e >> 4, kp = e & 15;
      const float2 xv = X2[(long)(m0 + r) * (D_MODEL / 2) + (k0 >> 1) + kp];
      As[r * 40 + 2 * kp]     = (_Float16)xv.x;
      As[r * 40 + 2 * kp + 1] = (_Float16)xv.y;
    }
#pragma unroll
    for (int i = 0; i < 4; ++i) {            // B: 64x32 f32 -> f16
      const int e = tid + i * 256;
      const int r = e >> 4, kp = e & 15;
      const float2 wv = W2[(long)(n0 + r) * (D_MODEL / 2) + (k0 >> 1) + kp];
      Bs[r * 40 + 2 * kp]     = (_Float16)wv.x;
      Bs[r * 40 + 2 * kp + 1] = (_Float16)wv.y;
    }
    __syncthreads();

    v16h a;
#pragma unroll
    for (int v = 0; v < 8; ++v) {
      const int ka = (v < 4) ? (kbA + 2 * v) : (16 + kbA + 2 * (v - 4));
      a[2 * v]     = As[(wave * 16 + row) * 40 + ka];
      a[2 * v + 1] = As[(wave * 16 + row) * 40 + ka + 1];
    }
#pragma unroll
    for (int t = 0; t < 4; ++t) {
      v16h b;
#pragma unroll
      for (int v = 0; v < 8; ++v) {
        const int kb = (v < 4) ? (kbA + 2 * v) : (16 + kbA + 2 * (v - 4));
        b[2 * v]     = Bs[(t * 16 + col) * 40 + kb];
        b[2 * v + 1] = Bs[(t * 16 + col) * 40 + kb + 1];
      }
      acc[t] = wmma_f16(a, b, acc[t]);
    }
  }

  const int mbase = m0 + wave * 16 + (lane >> 4) * 8;  // 8 consecutive rows/lane
  const int b_ = mbase >> 12;                 // batch (uniform over the 8 rows)
  const int sbase = mbase & (SEQ - 1);
#pragma unroll
  for (int t = 0; t < 4; ++t) {
    const int n  = n0 + t * 16 + col;
    const float bn = bias[n];
    const int h  = n >> 6, dk = n & 63;
    if (vmode == 0) {
#pragma unroll
      for (int r = 0; r < 8; ++r)
        Out[(((long)(b_ * HEADS + h) * SEQ + (sbase + r)) * DK) + dk] =
            (_Float16)(acc[t][r] + bn);
    } else {
      v8h pk;
#pragma unroll
      for (int r = 0; r < 8; ++r) pk[r] = (_Float16)(acc[t][r] + bn);
      *(v8h*)(Out + ((long)((b_ * HEADS + h) * DK + dk)) * SEQ + sbase) = pk;
    }
  }
}

// ---------------------------------------------------------------------------
// Kernel 2: flash attention. Block = 8 waves, one (b,h); wave owns a 16-row
// query strip. K[key][d] and Vt[d][key] 64-key tiles double-buffered in LDS;
// TDM path: wave 0 issues next tile's 2 DMA ops, s_wait_tensorcnt(2) retires
// the current tile while the next is in flight. Padded 72-half rows -> all
// B/A fragments are contiguous 16B-aligned ds_load_b128.
// ---------------------------------------------------------------------------
__global__ __launch_bounds__(256) void flash_attn_kernel(
    const _Float16* __restrict__ Qh, const _Float16* __restrict__ Kh,
    const _Float16* __restrict__ Vt, _Float16* __restrict__ Att)
{
  __shared__ _Float16 ldsK[2][64 * 72];       // [key][d]
  __shared__ _Float16 ldsV[2][64 * 72];       // [d][key]
  __shared__ _Float16 ldsP[8][16 * 72];       // per-wave P tile

  const int tid   = threadIdx.x;
  const int lane  = tid & 31;
  const int wave  = tid >> 5;
  const int task  = blockIdx.x * 8 + wave;    // 16 bh * 256 strips
  const int bh    = task >> 8;                // uniform across the block
  const int strip = task & 255;
  const long baseK = (long)bh * SEQ * DK;     // K/Q base (row-major)
  const long baseV = (long)bh * DK * SEQ;     // Vt base (dim-major)

  const int row = lane & 15;
  const int kbA = (lane >> 4) * 8;
  const int kbB = (lane >> 4) * 16;
  const int col = lane & 15;
  _Float16* P = &ldsP[wave][0];

  // Q strip -> two A-fragments
  const _Float16* __restrict__ qrow = Qh + baseK + (long)(strip * 16 + row) * DK;
  v16h qa0, qa1;
#pragma unroll
  for (int v = 0; v < 8; ++v) {
    const int k = (v < 4) ? (kbA + 2 * v) : (16 + kbA + 2 * (v - 4));
    qa0[2 * v] = qrow[k];      qa0[2 * v + 1] = qrow[k + 1];
    qa1[2 * v] = qrow[32 + k]; qa1[2 * v + 1] = qrow[32 + k + 1];
  }

  v8f o[4];
#pragma unroll
  for (int t = 0; t < 4; ++t) o[t] = (v8f){};
  float m_i[8], l_i[8];
#pragma unroll
  for (int r = 0; r < 8; ++r) { m_i[r] = -3.0e38f; l_i[r] = 0.f; }

#ifdef HAVE_TDM
  if (wave == 0) {   // prologue: tile 0 -> buffer 0
    tdm_load_2d((unsigned int)(unsigned long long)(void*)ldsK[0],
                Kh + baseK, 32, 32, 64, SEQ, 32, 4, 3, 1);
    tdm_load_2d((unsigned int)(unsigned long long)(void*)ldsV[0],
                Vt + baseV, 2048, 32, 64, DK, 2048, 4, 3, 1);
  }
#endif

  int cur = 0;
  for (int j = 0; j < SEQ; j += 64, cur ^= 1) {
    const int has_next = (j + 64 < SEQ);
#ifdef HAVE_TDM
    if (wave == 0) {
      if (has_next) {   // issue tile i+1 into the other buffer
        tdm_load_2d((unsigned int)(unsigned long long)(void*)ldsK[cur ^ 1],
                    Kh + baseK + (long)(j + 64) * DK,
                    32, 32, 64, SEQ, 32, 4, 3, 1);
        tdm_load_2d((unsigned int)(unsigned long long)(void*)ldsV[cur ^ 1],
                    Vt + baseV + (j + 64),
                    2048, 32, 64, DK, 2048, 4, 3, 1);
        __builtin_amdgcn_s_wait_tensorcnt(2);   // tile i retired, i+1 in flight
      } else {
        __builtin_amdgcn_s_wait_tensorcnt(0);
      }
    }
#else
    {
      const unsigned int* __restrict__ Ksrc =
          (const unsigned int*)(Kh + baseK + (long)j * DK);
      const unsigned int* __restrict__ Vsrc32 = (const unsigned int*)Vt;
      unsigned int* Kdst = (unsigned int*)ldsK[cur];
      unsigned int* Vdst = (unsigned int*)ldsV[cur];
#pragma unroll
      for (int i = 0; i < 8; ++i) {            // 2048 dw per tile / 256 thr
        const int e = tid + i * 256;
        const int r = e >> 5, kd = e & 31;     // 32 dw per row
        Kdst[r * 36 + kd] = Ksrc[r * 32 + kd];
        Vdst[r * 36 + kd] =
            Vsrc32[(baseV >> 1) + (long)r * (SEQ / 2) + (j >> 1) + kd];
      }
      if (has_next) {
        __builtin_prefetch(Kh + baseK + (long)(j + 64 + col) * DK, 0, 0);
        __builtin_prefetch(Vt + baseV + (long)col * SEQ + j + 64, 0, 0);
      }
    }
#endif
    __syncthreads();                           // tile i visible to all waves
    const _Float16* __restrict__ kbuf = ldsK[cur];
    const _Float16* __restrict__ vbuf = ldsV[cur];

    // ---- scores: 4 key sub-tiles of 16 keys, 2 WMMA each -------------------
    v8f sc[4];
#pragma unroll
    for (int t = 0; t < 4; ++t) {
      const _Float16* __restrict__ krow = kbuf + (t * 16 + col) * 72;
      v16h b0, b1;
#pragma unroll
      for (int v = 0; v < 8; ++v) {
        const int d = (v < 4) ? (kbA + 2 * v) : (16 + kbA + 2 * (v - 4));
        b0[2 * v] = krow[d];      b0[2 * v + 1] = krow[d + 1];
        b1[2 * v] = krow[32 + d]; b1[2 * v + 1] = krow[32 + d + 1];
      }
      v8f c = {};
      c = wmma_f16(qa0, b0, c);
      c = wmma_f16(qa1, b1, c);
      sc[t] = c * 0.125f;                      // 1/sqrt(64)
    }

    // ---- online softmax ----------------------------------------------------
#pragma unroll
    for (int r = 0; r < 8; ++r) {
      float v = fmaxf(fmaxf(sc[0][r], sc[1][r]), fmaxf(sc[2][r], sc[3][r]));
      v = fmaxf(v, __shfl_xor(v, 1, 32));
      v = fmaxf(v, __shfl_xor(v, 2, 32));
      v = fmaxf(v, __shfl_xor(v, 4, 32));
      v = fmaxf(v, __shfl_xor(v, 8, 32));
      const float mnew  = fmaxf(m_i[r], v);
      const float alpha = __expf(m_i[r] - mnew);
      float s0 = 0.f;
#pragma unroll
      for (int t = 0; t < 4; ++t) {
        const float p = __expf(sc[t][r] - mnew);
        sc[t][r] = p;
        s0 += p;
      }
      s0 += __shfl_xor(s0, 1, 32);
      s0 += __shfl_xor(s0, 2, 32);
      s0 += __shfl_xor(s0, 4, 32);
      s0 += __shfl_xor(s0, 8, 32);
      l_i[r] = l_i[r] * alpha + s0;
      m_i[r] = mnew;
#pragma unroll
      for (int t = 0; t < 4; ++t) o[t][r] *= alpha;
    }

    // ---- P: C-layout -> per-wave LDS tile ----------------------------------
    {
      const int mb = (lane >> 4) * 8;
#pragma unroll
      for (int t = 0; t < 4; ++t)
#pragma unroll
        for (int r = 0; r < 8; ++r)
          P[(mb + r) * 72 + t * 16 + col] = (_Float16)sc[t][r];
    }

    // ---- reload P as A-fragments ------------------------------------------
    v16h pa0, pa1;
#pragma unroll
    for (int v = 0; v < 8; ++v) {
      const int k = (v < 4) ? (kbA + 2 * v) : (16 + kbA + 2 * (v - 4));
      pa0[2 * v] = P[row * 72 + k];      pa0[2 * v + 1] = P[row * 72 + k + 1];
      pa1[2 * v] = P[row * 72 + 32 + k]; pa1[2 * v + 1] = P[row * 72 + 32 + k + 1];
    }

    // ---- O += P(16x64) * V(64x64) ------------------------------------------
#pragma unroll
    for (int t = 0; t < 4; ++t) {
      const _Float16* __restrict__ vrow = vbuf + (t * 16 + col) * 72;
      v16h vb0, vb1;
#pragma unroll
      for (int v = 0; v < 8; ++v) {
        const int k = (v < 4) ? (kbA + 2 * v) : (16 + kbA + 2 * (v - 4));
        vb0[2 * v] = vrow[k];      vb0[2 * v + 1] = vrow[k + 1];
        vb1[2 * v] = vrow[32 + k]; vb1[2 * v + 1] = vrow[32 + k + 1];
      }
      o[t] = wmma_f16(pa0, vb0, o[t]);
      o[t] = wmma_f16(pa1, vb1, o[t]);
    }
    __syncthreads();   // everyone done with buf[cur] before it is re-filled
  }

  // ---- epilogue --------------------------------------------------------------
  const int mb = (lane >> 4) * 8;
  const int b_ = bh >> 3;
  const int h  = bh & 7;
#pragma unroll
  for (int r = 0; r < 8; ++r) {
    const float inv = 1.0f / l_i[r];
    const int s = strip * 16 + mb + r;
#pragma unroll
    for (int t = 0; t < 4; ++t)
      Att[((long)(b_ * SEQ + s)) * D_MODEL + h * 64 + t * 16 + col] =
          (_Float16)(o[t][r] * inv);
  }
}

// ---------------------------------------------------------------------------
// Kernel 3: output projection  out_f32 = attended_f16 @ Wo^T + b_o
// ---------------------------------------------------------------------------
__global__ __launch_bounds__(256) void out_proj_kernel(
    const _Float16* __restrict__ A, const float* __restrict__ W,
    const float* __restrict__ bias, float* __restrict__ Out)
{
  __shared__ _Float16 As[128 * 40];
  __shared__ _Float16 Bs[64 * 40];

  const int tid  = threadIdx.x;
  const int lane = tid & 31;
  const int wave = tid >> 5;
  const int m0 = (blockIdx.x >> 3) * 128;
  const int n0 = (blockIdx.x & 7) * 64;
  const int row = lane & 15;
  const int kbA = (lane >> 4) * 8;
  const int kbB = (lane >> 4) * 16;
  const int col = lane & 15;

  v8f acc[4];
#pragma unroll
  for (int t = 0; t < 4; ++t) acc[t] = (v8f){};

  const unsigned int* __restrict__ A32 = (const unsigned int*)A;
  const float2* __restrict__ W2 = (const float2*)W;

  for (int k0 = 0; k0 < D_MODEL; k0 += 32) {
    __syncthreads();
    unsigned int* As32 = (unsigned int*)As;
#pragma unroll
    for (int i = 0; i < 8; ++i) {
      const int e = tid + i * 256;
      const int r = e >> 4, kd = e & 15;      // 16 dw per row
      As32[r * 20 + kd] = A32[(long)(m0 + r) * (D_MODEL / 2) + (k0 >> 1) + kd];
    }
#pragma unroll
    for (int i = 0; i < 4; ++i) {
      const int e = tid + i * 256;
      const int r = e >> 4, kp = e & 15;
      const float2 wv = W2[(long)(n0 + r) * (D_MODEL / 2) + (k0 >> 1) + kp];
      Bs[r * 40 + 2 * kp]     = (_Float16)wv.x;
      Bs[r * 40 + 2 * kp + 1] = (_Float16)wv.y;
    }
    __syncthreads();

    v16h a;
#pragma unroll
    for (int v = 0; v < 8; ++v) {
      const int ka = (v < 4) ? (kbA + 2 * v) : (16 + kbA + 2 * (v - 4));
      a[2 * v]     = As[(wave * 16 + row) * 40 + ka];
      a[2 * v + 1] = As[(wave * 16 + row) * 40 + ka + 1];
    }
#pragma unroll
    for (int t = 0; t < 4; ++t) {
      v16h b;
#pragma unroll
      for (int v = 0; v < 8; ++v) {
        const int kb = (v < 4) ? (kbA + 2 * v) : (16 + kbA + 2 * (v - 4));
        b[2 * v]     = Bs[(t * 16 + col) * 40 + kb];
        b[2 * v + 1] = Bs[(t * 16 + col) * 40 + kb + 1];
      }
      acc[t] = wmma_f16(a, b, acc[t]);
    }
  }

  const int mbase = m0 + wave * 16 + (lane >> 4) * 8;
#pragma unroll
  for (int t = 0; t < 4; ++t) {
    const float bn = bias[n0 + t * 16 + col];
#pragma unroll
    for (int r = 0; r < 8; ++r)
      Out[(long)(mbase + r) * D_MODEL + n0 + t * 16 + col] = acc[t][r] + bn;
  }
}

// ---------------------------------------------------------------------------
extern "C" void kernel_launch(void* const* d_in, const int* in_sizes, int n_in,
                              void* d_out, int out_size, void* d_ws, size_t ws_size,
                              hipStream_t stream) {
  const float* query = (const float*)d_in[0];
  const float* key_  = (const float*)d_in[1];
  const float* value = (const float*)d_in[2];
  const float* w_q   = (const float*)d_in[3];
  const float* b_q   = (const float*)d_in[4];
  const float* w_k   = (const float*)d_in[5];
  const float* b_k   = (const float*)d_in[6];
  const float* w_v   = (const float*)d_in[7];
  const float* b_v   = (const float*)d_in[8];
  const float* w_o   = (const float*)d_in[9];
  const float* b_o   = (const float*)d_in[10];

  const size_t matElems = (size_t)M_TOT * D_MODEL;   // 4,194,304 f16 each
  _Float16* Qh  = (_Float16*)d_ws;
  _Float16* Kh  = Qh + matElems;
  _Float16* Vt  = Kh + matElems;                     // [B][H][DK][S]
  _Float16* Att = Vt + matElems;                     // total 32 MB of ws

  dim3 blk(256);
  qkv_proj_kernel<<<512, blk, 0, stream>>>(query, w_q, b_q, Qh, 0);
  qkv_proj_kernel<<<512, blk, 0, stream>>>(key_,  w_k, b_k, Kh, 0);
  qkv_proj_kernel<<<512, blk, 0, stream>>>(value, w_v, b_v, Vt, 1);
  flash_attn_kernel<<<512, blk, 0, stream>>>(Qh, Kh, Vt, Att);
  out_proj_kernel<<<512, blk, 0, stream>>>(Att, w_o, b_o, (float*)d_out);
}
